// QuantQKVProj_35098472742916
// MI455X (gfx1250) — 8-bit matrix operand order repaired and checked by reading, compiled, not run on hardware
//
#include <hip/hip_runtime.h>

typedef __attribute__((ext_vector_type(8))) int v8i;
typedef int v4i __attribute__((ext_vector_type(4)));
typedef __attribute__((address_space(1))) v4i* gptr_v4i;
typedef __attribute__((address_space(3))) v4i* lptr_v4i;

#define TILE_M 128
#define TILE_N 128
#define TILE_K 64

#if defined(__AMDGCN__) && \
    __has_builtin(__builtin_amdgcn_global_load_async_to_lds_b128) && \
    __has_builtin(__builtin_amdgcn_s_wait_asynccnt)
#define USE_ASYNC_LDS 1
#else
#define USE_ASYNC_LDS 0
#endif

__global__ void quant_i8_kernel(const float* __restrict__ x,
                                const float* __restrict__ scale,
                                unsigned int* __restrict__ out, int n4) {
    int i = blockIdx.x * blockDim.x + threadIdx.x;
    if (i >= n4) return;
    const float inv = 1.0f / scale[0];
    float4 v = reinterpret_cast<const float4*>(x)[i];
    int q0 = (int)fminf(fmaxf(rintf(v.x * inv), -128.0f), 127.0f);
    int q1 = (int)fminf(fmaxf(rintf(v.y * inv), -128.0f), 127.0f);
    int q2 = (int)fminf(fmaxf(rintf(v.z * inv), -128.0f), 127.0f);
    int q3 = (int)fminf(fmaxf(rintf(v.w * inv), -128.0f), 127.0f);
    out[i] = (unsigned int)(q0 & 0xFF) | ((unsigned int)(q1 & 0xFF) << 8) |
             ((unsigned int)(q2 & 0xFF) << 16) | ((unsigned int)(q3 & 0xFF) << 24);
}

__global__ __launch_bounds__(256, 2)
void qkv_gemm_iu8_kernel(const unsigned char* __restrict__ Aq,
                         const unsigned char* __restrict__ Wq,
                         const float* __restrict__ bias,
                         const float* __restrict__ s_in,
                         const float* __restrict__ s_w,
                         const float* __restrict__ s_out,
                         float* __restrict__ out,
                         int M, int N, int K, int D) {
    __shared__ int4 sA4[TILE_M * TILE_K / 16];
    __shared__ int4 sB4[TILE_N * TILE_K / 16];

    const int tid    = threadIdx.x;
    const int wave   = tid >> 5;
    const int lane   = tid & 31;
    const int laneLo = lane & 15;
    const int laneHi = lane >> 4;

    const int mBlock = blockIdx.y * TILE_M;
    const int nBlock = blockIdx.x * TILE_N;
    const int wm = (wave >> 1) * 32;
    const int wn = (wave & 1) * 64;

    v8i acc[2][4];
#pragma unroll
    for (int i = 0; i < 2; ++i)
#pragma unroll
        for (int j = 0; j < 4; ++j)
#pragma unroll
            for (int r = 0; r < 8; ++r) acc[i][j][r] = 0;

    const unsigned char* sA = reinterpret_cast<const unsigned char*>(sA4);
    const unsigned char* sB = reinterpret_cast<const unsigned char*>(sB4);

    for (int k0 = 0; k0 < K; k0 += TILE_K) {
#if USE_ASYNC_LDS
#pragma unroll
        for (int i = 0; i < 2; ++i) {
            int c   = tid + i * 256;
            int row = c >> 2;
            int col = (c & 3) * 16;
            __builtin_amdgcn_global_load_async_to_lds_b128(
                (gptr_v4i)(Aq + (size_t)(mBlock + row) * K + k0 + col),
                (lptr_v4i)&sA4[c], 0, 0);
            __builtin_amdgcn_global_load_async_to_lds_b128(
                (gptr_v4i)(Wq + (size_t)(nBlock + row) * K + k0 + col),
                (lptr_v4i)&sB4[c], 0, 0);
        }
        __builtin_amdgcn_s_wait_asynccnt(0);
#else
#pragma unroll
        for (int i = 0; i < 2; ++i) {
            int c   = tid + i * 256;
            int row = c >> 2;
            int col = (c & 3) * 16;
            sA4[c] = *reinterpret_cast<const int4*>(
                Aq + (size_t)(mBlock + row) * K + k0 + col);
            sB4[c] = *reinterpret_cast<const int4*>(
                Wq + (size_t)(nBlock + row) * K + k0 + col);
        }
#endif
        __syncthreads();

        v8i afrag[2], bfrag[4];
#pragma unroll
        for (int im = 0; im < 2; ++im) {
            const unsigned char* ap =
                sA + (size_t)(wm + im * 16 + laneLo) * TILE_K + laneHi * 16;
            int4 p0 = *reinterpret_cast<const int4*>(ap +  0);
            int4 p1 = *reinterpret_cast<const int4*>(ap + 32);
            v8i a;
            a[0] = p0.x; a[1] = p0.y; a[2] = p0.z; a[3] = p0.w;
            a[4] = p1.x; a[5] = p1.y; a[6] = p1.z; a[7] = p1.w;
            afrag[im] = a;
        }
#pragma unroll
        for (int in = 0; in < 4; ++in) {
            const unsigned char* bp =
                sB + (size_t)(wn + in * 16 + laneLo) * TILE_K + laneHi * 16;
            int4 q0 = *reinterpret_cast<const int4*>(bp +  0);
            int4 q1 = *reinterpret_cast<const int4*>(bp + 32);
            v8i b;
            b[0] = q0.x; b[1] = q0.y; b[2] = q0.z; b[3] = q0.w;
            b[4] = q1.x; b[5] = q1.y; b[6] = q1.z; b[7] = q1.w;
            bfrag[in] = b;
        }

#pragma unroll
        for (int im = 0; im < 2; ++im)
#pragma unroll
            for (int in = 0; in < 4; ++in)
                acc[im][in] = __builtin_amdgcn_wmma_i32_16x16x64_iu8(
                    true, afrag[im], true, bfrag[in], acc[im][in], false, false);

        __syncthreads();
    }

    const float sisw   = s_in[0] * s_w[0];
    const float so     = s_out[0];
    const float inv_so = 1.0f / so;
    const size_t MD    = (size_t)M * (size_t)D;

#pragma unroll
    for (int im = 0; im < 2; ++im) {
#pragma unroll
        for (int in = 0; in < 4; ++in) {
            const int nCol = nBlock + wn + in * 16 + laneLo;
            const int part = nCol / D;
            const int col  = nCol % D;
            const float bv = bias[nCol];
            float* outBase = out + (size_t)part * MD + col;
#pragma unroll
            for (int r = 0; r < 8; ++r) {
                const int mRow = mBlock + wm + im * 16 + r + laneHi * 8;
                float raw = (float)acc[im][in][r] * sisw + bv;
                float q = fminf(fmaxf(rintf(raw * inv_so), -128.0f), 127.0f) * so;
                outBase[(size_t)mRow * D] = q;
            }
        }
    }
}

extern "C" void kernel_launch(void* const* d_in, const int* in_sizes, int n_in,
                              void* d_out, int out_size, void* d_ws, size_t ws_size,
                              hipStream_t stream) {
    const float* x    = (const float*)d_in[0];
    const float* W    = (const float*)d_in[1];
    const float* b    = (const float*)d_in[2];
    const float* s_in = (const float*)d_in[3];
    const float* s_w  = (const float*)d_in[4];
    const float* s_out= (const float*)d_in[5];
    float* out = (float*)d_out;

    const int D = in_sizes[2] / 3;
    const int K = D;
    const int N = 3 * D;
    const int M = in_sizes[0] / D;

    unsigned char* Aq = (unsigned char*)d_ws;
    unsigned char* Wq = Aq + (size_t)M * (size_t)K;

    {
        int n4 = (M * K) / 4;
        quant_i8_kernel<<<(n4 + 255) / 256, 256, 0, stream>>>(
            x, s_in, (unsigned int*)Aq, n4);
    }
    {
        int n4 = (N * K) / 4;
        quant_i8_kernel<<<(n4 + 255) / 256, 256, 0, stream>>>(
            W, s_w, (unsigned int*)Wq, n4);
    }

    dim3 grid(N / TILE_N, M / TILE_M);
    qkv_gemm_iu8_kernel<<<grid, 256, 0, stream>>>(
        Aq, Wq, b, s_in, s_w, s_out, out, M, N, K, D);
}
